// ManhattanSelfAttention_16441134809684
// MI455X (gfx1250) — matrix operand order repaired and checked by reading, not compiled, not run on hardware
//
#include <hip/hip_runtime.h>
#include <hip/hip_bf16.h>

// MaSA decomposed attention for MI455X (gfx1250, wave32, WMMA).
// B=8, C=d=256, H=W=128, L=128 both passes. All matrix math in
// v_wmma_f32_16x16x32_bf16 (bf16 in, f32 accum); softmax/decay in fp32 VALU.
// K matrix staged to LDS via gfx1250 async load-to-LDS when available.

typedef __attribute__((ext_vector_type(16))) __bf16 v16bf;
typedef __attribute__((ext_vector_type(8)))  float  v8f;
typedef __attribute__((ext_vector_type(4)))  int    v4i;

typedef v4i __attribute__((address_space(1)))* gptr_v4i;   // global
typedef v4i __attribute__((address_space(3)))* lptr_v4i;   // LDS

union ABFrag {
    v16bf v;
    uint4 q[2];
};

#define NPOS   131072        // B*H*W
#define DIM    256
#define SEQL   128

#if __has_builtin(__builtin_amdgcn_global_load_async_to_lds_b128) && \
    __has_builtin(__builtin_amdgcn_s_wait_asynccnt)
#define USE_ASYNC_LDS 1
#else
#define USE_ASYNC_LDS 0
#endif

// ---------------------------------------------------------------------------
// Stage 0: x [B,C,H,W] f32 -> Xh [pos=(b*H+h)*W+w, c] bf16
// ---------------------------------------------------------------------------
__global__ __launch_bounds__(256) void pack_x_kernel(const float* __restrict__ x,
                                                     __bf16* __restrict__ Xh) {
    long n = blockIdx.x;                  // position
    int  c = threadIdx.x;
    int  b = (int)(n >> 14);
    int  h = (int)((n >> 7) & 127);
    int  w = (int)(n & 127);
    float v = x[((((long)b * DIM + c) * 128 + h) << 7) + w];
    Xh[n * DIM + c] = (__bf16)v;
}

// ---------------------------------------------------------------------------
// Stage 0b: weights f32 [256,256] -> bf16, 3 matrices packed
// ---------------------------------------------------------------------------
__global__ __launch_bounds__(256) void pack_w_kernel(const float* __restrict__ Wq,
                                                     const float* __restrict__ Wk,
                                                     const float* __restrict__ Wv,
                                                     __bf16* __restrict__ Wb) {
    int p = blockIdx.y;                   // 0=q 1=k 2=v
    int j = blockIdx.x;                   // output channel
    int k = threadIdx.x;
    const float* Ws = (p == 0) ? Wq : ((p == 1) ? Wk : Wv);
    Wb[(((long)p * DIM + j) * DIM) + k] = (__bf16)Ws[(long)j * DIM + k];
}

// ---------------------------------------------------------------------------
// Stage 1: Y_p = X @ W_p^T + b_p   (p in {q,k,v}, fused via blockIdx.y)
// block = 128 threads = 4 waves; wave computes a 32(M) x 64(N) tile
// (2 M-tiles x 4 N-tiles = 8 accumulators -> 8 independent WMMAs per k-step).
// ---------------------------------------------------------------------------
__global__ __launch_bounds__(128) void gemm_qkv_kernel(const __bf16* __restrict__ X,
                                                       const __bf16* __restrict__ Wb,
                                                       const float* __restrict__ bq,
                                                       const float* __restrict__ bk,
                                                       const float* __restrict__ bv,
                                                       __bf16* __restrict__ Qo,
                                                       __bf16* __restrict__ Ko,
                                                       __bf16* __restrict__ Vo) {
    int p = blockIdx.y;
    const __bf16* Wp   = Wb + (long)p * DIM * DIM;
    const float*  bias = (p == 0) ? bq : ((p == 1) ? bk : bv);
    __bf16*       Yo   = (p == 0) ? Qo : ((p == 1) ? Ko : Vo);

    int  lane = threadIdx.x & 31;
    int  wave = threadIdx.x >> 5;
    long m0   = (long)blockIdx.x * 32;    // 2 M-tiles per wave
    int  n0   = wave * 64;

    int  lh   = lane >> 4;                 // lane half
    int  akh  = lh << 3;                   // A per-lane K sub-chunk: 0 or 8
    int  bkh  = lh << 3;                   // B per-lane K sub-chunk: 0 or 8
    long arow0 = m0 + (lane & 15);
    long arow1 = arow0 + 16;

    v8f acc[2][4] = {};
    for (int k0 = 0; k0 < DIM; k0 += 32) {
        const __bf16* ap0 = X + arow0 * DIM + k0 + akh;
        const __bf16* ap1 = X + arow1 * DIM + k0 + akh;
        if (k0 + 32 < DIM) __builtin_prefetch(ap0 + 32, 0, 1);   // global_prefetch_b8
        ABFrag a0, a1;
        a0.q[0] = *(const uint4*)ap0;            // K = k0+akh .. +7
        a0.q[1] = *(const uint4*)(ap0 + 16);     // K = k0+16+akh .. +7
        a1.q[0] = *(const uint4*)ap1;
        a1.q[1] = *(const uint4*)(ap1 + 16);
        for (int j = 0; j < 4; ++j) {
            int ncol = n0 + j * 16 + (lane & 15);
            const __bf16* bp = Wp + (long)ncol * DIM + k0 + bkh;
            ABFrag b;
            b.q[0] = *(const uint4*)bp;          // K = k0+bkh .. +7
            b.q[1] = *(const uint4*)(bp + 16);
            acc[0][j] = __builtin_amdgcn_wmma_f32_16x16x32_bf16(
                false, a0.v, false, b.v, (short)0, acc[0][j], false, false);
            acc[1][j] = __builtin_amdgcn_wmma_f32_16x16x32_bf16(
                false, a1.v, false, b.v, (short)0, acc[1][j], false, false);
        }
    }
    for (int t = 0; t < 2; ++t) {
        for (int j = 0; j < 4; ++j) {
            for (int rr = 0; rr < 8; ++rr) {
                long m = m0 + t * 16 + rr + (lh << 3);   // lanes 16-31 hold M=8..15
                int  n = n0 + j * 16 + (lane & 15);
                Yo[m * DIM + n] = (__bf16)(acc[t][j][rr] + bias[n]);
            }
        }
    }
}

// ---------------------------------------------------------------------------
// Stage 2: per-sequence attention with post-softmax Manhattan decay.
// block = 256 threads = 8 waves, one sequence of L=128 per block.
// pass 0: seq along W (step=1); pass 1: seq along H (step=128 positions).
// K staged once to LDS (async when available); V^T staged per 32-key chunk.
// ---------------------------------------------------------------------------
__global__ __launch_bounds__(256) void attn_kernel(const __bf16* __restrict__ Qb,
                                                   const __bf16* __restrict__ Kb,
                                                   const __bf16* __restrict__ Vb,
                                                   __bf16* __restrict__ O,
                                                   int pass) {
    __shared__ __align__(16) __bf16 K_lds[SEQL][DIM];    // 64 KB keys
    __shared__ __align__(16) __bf16 P_lds[SEQL][SEQL];   // 32 KB probs
    __shared__ __align__(16) __bf16 Vt_lds[DIM][32];     // 16 KB V^T chunk

    int r = blockIdx.x;                   // sequence id, 0..1023
    long base;
    int  step;
    if (pass == 0) { base = (long)r << 7; step = 1; }
    else           { int b = r >> 7, w = r & 127; base = ((long)b << 14) + w; step = 128; }

    int lane = threadIdx.x & 31;
    int wave = threadIdx.x >> 5;
    int m0   = wave * 16;
    int lh   = lane >> 4;
    int akh  = lh << 3;
    int bkh  = lh << 3;

    // ---- stage K[seq] -> LDS (row-major), async-to-LDS when available ----
    for (int idx = threadIdx.x; idx < (SEQL * DIM) / 8; idx += 256) {
        int key = idx >> 5;               // 32 8-elem chunks per row
        int c8  = (idx & 31) << 3;
        const __bf16* g = Kb + (base + (long)key * step) * DIM + c8;
#if USE_ASYNC_LDS
        __builtin_amdgcn_global_load_async_to_lds_b128(
            (gptr_v4i)g, (lptr_v4i)&K_lds[key][c8], 0, 0);
#else
        *(uint4*)&K_lds[key][c8] = *(const uint4*)g;
#endif
    }
#if USE_ASYNC_LDS
    __builtin_amdgcn_s_wait_asynccnt(0);
#endif
    __syncthreads();

    // ---- S = Q K^T / sqrt(d) ----
    v8f sacc[8] = {};
    long qpos = base + (long)(m0 + (lane & 15)) * step;
    for (int d0 = 0; d0 < DIM; d0 += 32) {
        ABFrag a;
        const __bf16* ap = Qb + qpos * DIM + d0 + akh;
        a.q[0] = *(const uint4*)ap;
        a.q[1] = *(const uint4*)(ap + 16);
        for (int j = 0; j < 8; ++j) {
            int key = j * 16 + (lane & 15);
            const __bf16* bp = &K_lds[key][d0 + bkh];
            ABFrag b;
            b.q[0] = *(const uint4*)bp;
            b.q[1] = *(const uint4*)(bp + 16);
            sacc[j] = __builtin_amdgcn_wmma_f32_16x16x32_bf16(
                false, a.v, false, b.v, (short)0, sacc[j], false, false);
        }
    }

    // ---- softmax per query row + gamma^|i-j| decay, probs -> LDS ----
    const float lng = -0.10536051565782628f;   // ln(0.9)
    for (int rr = 0; rr < 8; ++rr) {
        int mloc = m0 + rr + (lh << 3);
        float mx = -1e30f;
        for (int j = 0; j < 8; ++j) mx = fmaxf(mx, sacc[j][rr] * 0.0625f);
        for (int o = 1; o < 16; o <<= 1) mx = fmaxf(mx, __shfl_xor(mx, o, 32));
        float e[8], sum = 0.f;
        for (int j = 0; j < 8; ++j) { e[j] = __expf(sacc[j][rr] * 0.0625f - mx); sum += e[j]; }
        for (int o = 1; o < 16; o <<= 1) sum += __shfl_xor(sum, o, 32);
        float inv = 1.0f / sum;
        for (int j = 0; j < 8; ++j) {
            int col = j * 16 + (lane & 15);
            float dec = __expf(lng * fabsf((float)(mloc - col)));
            P_lds[mloc][col] = (__bf16)(e[j] * inv * dec);
        }
    }

    // ---- O = P @ V ----
    v8f oacc[16] = {};
    for (int kc = 0; kc < 4; ++kc) {           // key chunks of 32
        __syncthreads();                        // Vt reuse WAR
        for (int idx = threadIdx.x; idx < 32 * DIM; idx += 256) {
            int kk = idx >> 8;                  // key within chunk
            int dd = idx & 255;                 // channel
            Vt_lds[dd][kk] = Vb[(base + (long)(kc * 32 + kk) * step) * DIM + dd];
        }
        __syncthreads();
        ABFrag a;                               // A = P[m0..m0+15][kc*32..+31]
        const __bf16* ap = &P_lds[m0 + (lane & 15)][kc * 32 + akh];
        a.q[0] = *(const uint4*)ap;
        a.q[1] = *(const uint4*)(ap + 16);
        for (int j = 0; j < 16; ++j) {          // 256 output channels
            int dd = j * 16 + (lane & 15);
            const __bf16* bp = &Vt_lds[dd][bkh];
            ABFrag b;
            b.q[0] = *(const uint4*)bp;
            b.q[1] = *(const uint4*)(bp + 16);
            oacc[j] = __builtin_amdgcn_wmma_f32_16x16x32_bf16(
                false, a.v, false, b.v, (short)0, oacc[j], false, false);
        }
    }
    for (int j = 0; j < 16; ++j) {
        for (int rr = 0; rr < 8; ++rr) {
            int m = m0 + rr + (lh << 3);
            int c = j * 16 + (lane & 15);
            O[(base + (long)m * step) * DIM + c] = (__bf16)(oacc[j][rr]);
        }
    }
}

// ---------------------------------------------------------------------------
// Stage 3: O2 bf16 [pos, c] -> out f32 [B,C,H,W]
// ---------------------------------------------------------------------------
__global__ __launch_bounds__(256) void unpack_kernel(const __bf16* __restrict__ O2,
                                                     float* __restrict__ out) {
    long n = blockIdx.x;
    int  c = threadIdx.x;
    int  b = (int)(n >> 14);
    int  h = (int)((n >> 7) & 127);
    int  w = (int)(n & 127);
    out[((((long)b * DIM + c) * 128 + h) << 7) + w] = (float)O2[n * DIM + c];
}

// ---------------------------------------------------------------------------
extern "C" void kernel_launch(void* const* d_in, const int* in_sizes, int n_in,
                              void* d_out, int out_size, void* d_ws, size_t ws_size,
                              hipStream_t stream) {
    const float* x  = (const float*)d_in[0];
    const float* Wq = (const float*)d_in[1];
    const float* bq = (const float*)d_in[2];
    const float* Wk = (const float*)d_in[3];
    const float* bk = (const float*)d_in[4];
    const float* Wv = (const float*)d_in[5];
    const float* bv = (const float*)d_in[6];
    float* out = (float*)d_out;

    // ws layout: [0,64MB) X/O ping buffer; Q/K/V 64MB each; weights bf16 384KB
    char* ws = (char*)d_ws;
    __bf16* XO = (__bf16*)(ws);
    __bf16* Qb = (__bf16*)(ws + (64ull  << 20));
    __bf16* Kb = (__bf16*)(ws + (128ull << 20));
    __bf16* Vb = (__bf16*)(ws + (192ull << 20));
    __bf16* Wb = (__bf16*)(ws + (256ull << 20));

    pack_w_kernel<<<dim3(DIM, 3), 256, 0, stream>>>(Wq, Wk, Wv, Wb);
    pack_x_kernel<<<NPOS, 256, 0, stream>>>(x, XO);

    // pass 1: attention along W
    gemm_qkv_kernel<<<dim3(NPOS / 32, 3), 128, 0, stream>>>(XO, Wb, bq, bk, bv, Qb, Kb, Vb);
    attn_kernel<<<1024, 256, 0, stream>>>(Qb, Kb, Vb, XO, 0);   // out_h -> XO

    // pass 2: attention along H (position-strided sequences, no transpose)
    gemm_qkv_kernel<<<dim3(NPOS / 32, 3), 128, 0, stream>>>(XO, Wb, bq, bk, bv, Qb, Kb, Vb);
    attn_kernel<<<1024, 256, 0, stream>>>(Qb, Kb, Vb, XO, 1);   // out_w -> XO

    unpack_kernel<<<NPOS, 256, 0, stream>>>(XO, out);
}
